// hDRMLPv8Embed_70076686402007
// MI455X (gfx1250) — compile-verified, matrix operand order checked by reading, not run on hardware
//
#include <hip/hip_runtime.h>
#include <math.h>

// ---------------------------------------------------------------------------
// CDNA5 (gfx1250) implementation of hDRMLPv8Embed forward.
// All dense convolutions -> im2col GEMM on v_wmma_f32_16x16x32_f16 (wave32).
// Activations stored fp16 between kernels (HBM-bandwidth optimal at 23.3TB/s),
// fp32 accumulation inside WMMA. All shape constants are template params so
// index math compiles to shifts/mads (no runtime idiv), and each wave
// register-blocks NTB output tiles to amortize the im2col A-gather.
// ---------------------------------------------------------------------------

typedef __attribute__((ext_vector_type(16))) _Float16 v16h;
typedef __attribute__((ext_vector_type(8)))  float    v8f;

__device__ __forceinline__ float gelu_f(float v) {
    // exact erf GELU (torch nn.GELU default)
    return 0.5f * v * (1.0f + erff(v * 0.70710678118654752f));
}

// K-index mapping for 16-bit A/B fragments (ISA 7.12.2, 16-bit A 16x32):
//  lanes 0-15 : VGPR0..3 -> K=0..7,  VGPR4..7 -> K=16..23
//  lanes 16-31: VGPR0..3 -> K=8..15, VGPR4..7 -> K=24..31
__device__ __host__ __forceinline__ constexpr int kmap(int e, int hi) {
    int p = e >> 1, w = e & 1;
    int base = hi ? 8 : 0;
    return (p < 4) ? (base + 2 * p + w) : (16 + base + 2 * (p - 4) + w);
}

// ---------------------------------------------------------------------------
// Weight packing: w is G batches of [OC x K] row-major fp32 matrices.
// Output layout: [G][NT][KC][lane 0..31][e 0..15] fp16, element =
//   w[g][oc = nt*16 + lane%16][k = kc*32 + kmap(e, lane>=16)], 0 if OOB.
// Consumer loads one v16h (32B) per lane -> exact B fragment.
// ---------------------------------------------------------------------------
__global__ __launch_bounds__(256)
void pack_w(const float* __restrict__ w, int G, int OC, int K,
            _Float16* __restrict__ out) {
    int NT = (OC + 15) >> 4;
    int KC = (K + 31) >> 5;
    size_t total = (size_t)G * NT * KC * 512;
    size_t id = (size_t)blockIdx.x * blockDim.x + threadIdx.x;
    if (id >= total) return;
    int e    = (int)(id & 15);
    int lane = (int)((id >> 4) & 31);
    size_t q = id >> 9;
    int kc = (int)(q % KC); q /= KC;
    int nt = (int)(q % NT);
    int gi = (int)(q / NT);
    int oc = nt * 16 + (lane & 15);
    int k  = kc * 32 + kmap(e, lane >> 4);
    float v = (oc < OC && k < K) ? w[((size_t)gi * OC + oc) * K + k] : 0.0f;
    out[id] = (_Float16)v;
}

__global__ __launch_bounds__(256)
void f32_to_f16(const float* __restrict__ in, _Float16* __restrict__ out, size_t n) {
    size_t id = (size_t)blockIdx.x * blockDim.x + threadIdx.x;
    if (id < n) out[id] = (_Float16)in[id];
}

// ---------------------------------------------------------------------------
// Generic WMMA conv: im2col GEMM, M = B*OH*OW (pixels), N = OC, K = C*KH*KW.
// One wave computes NTB adjacent 16x16 output tiles (shared A fragment).
// Epilogue fuses BN (+GELU) and writes either fp16 NCHW (TOOUT=false) or a
// fp32 transposed [B,196,768] slice of d_out (TOOUT=true).
// ---------------------------------------------------------------------------
template <int C, int H, int W, int OC, int KH, int KWt, int STR, int PAD,
          int OH, int OW, int NTB, bool TOOUT, bool GELU>
__global__ __launch_bounds__(256)
void conv_wmma(const _Float16* __restrict__ src,
               const _Float16* __restrict__ wpk,
               const float* __restrict__ scale, const float* __restrict__ bias,
               _Float16* __restrict__ dst16, float* __restrict__ dstf,
               int ch_off) {
    constexpr int B      = 16;
    constexpr int K      = C * KH * KWt;
    constexpr int KHW    = KH * KWt;
    constexpr int NT     = (OC + 15) / 16;
    constexpr int KC     = (K + 31) / 32;
    constexpr int NB     = NT / NTB;          // NTB must divide NT
    constexpr int OHW    = OH * OW;
    constexpr int Mtiles = B * OHW / 16;

    const int lane = threadIdx.x & 31;
    const int wave = blockIdx.x * (blockDim.x >> 5) + (threadIdx.x >> 5);
    if (wave >= Mtiles * NB) return;
    const int mtile = wave / NB;
    const int nblk  = wave - mtile * NB;

    // A-fragment row owned by this lane
    const int hi  = lane >> 4;
    const int m   = lane & 15;
    const int pix = mtile * 16 + m;
    const int b   = pix / OHW;                 // constant divisor
    const int rem = pix - b * OHW;
    const int oy  = rem / OW;
    const int ox  = rem - oy * OW;
    const int iy0 = oy * STR - PAD;
    const int ix0 = ox * STR - PAD;
    const _Float16* sb = src + (size_t)b * C * H * W;

    v8f acc[NTB] = {};
#pragma unroll 1
    for (int kc = 0; kc < KC; ++kc) {
        const _Float16* bp =
            wpk + (((size_t)nblk * NTB * KC + kc) * 32 + lane) * 16;
        if (kc + 1 < KC)
            __builtin_prefetch(bp + 512, 0, 0);
        v16h a = {};
#pragma unroll
        for (int e = 0; e < 16; ++e) {
            const int k = kc * 32 + kmap(e, hi);
            _Float16 v = (_Float16)0.0f;
            if ((K % 32 == 0) || (k < K)) {
                const int ic = k / KHW;        // constant divisors -> mads
                const int r  = k - ic * KHW;
                const int ky = r / KWt;
                const int kx = r - ky * KWt;
                const int iy = iy0 + ky;
                const int ix = ix0 + kx;
                if (PAD == 0 || (iy >= 0 && iy < H && ix >= 0 && ix < W))
                    v = sb[((size_t)ic * H + iy) * W + ix];
            }
            a[e] = v;
        }
#pragma unroll
        for (int j = 0; j < NTB; ++j) {
            v16h bf = *(const v16h*)(bp + (size_t)j * KC * 512);
            acc[j] = __builtin_amdgcn_wmma_f32_16x16x32_f16(
                false, a, false, bf, (short)0, acc[j], false, false);
        }
    }

    // C/D layout: lane holds (M = v + hi*8, N = lane&15)
    const int n = lane & 15;
#pragma unroll
    for (int j = 0; j < NTB; ++j) {
        const int oc = (nblk * NTB + j) * 16 + n;
        if ((OC % 16 != 0) && oc >= OC) continue;
        const float s  = scale[oc];
        const float bb = bias[oc];
#pragma unroll
        for (int v = 0; v < 8; ++v) {
            const int mm = v + hi * 8;
            const int pp = mtile * 16 + mm;
            const int b2 = pp / OHW;
            const int r2 = pp - b2 * OHW;
            float val = acc[j][v] * s + bb;
            if (GELU) val = gelu_f(val);
            if constexpr (TOOUT) {
                // branch output: [B,196,768] with channel-concat offset
                dstf[((size_t)b2 * OHW + r2) * 768 + ch_off + oc] = val;
            } else {
                const int oy2 = r2 / OW, ox2 = r2 - oy2 * OW;
                dst16[(((size_t)b2 * OC + oc) * OH + oy2) * OW + ox2] =
                    (_Float16)val;
            }
        }
    }
}

// ---------------------------------------------------------------------------
// RegionLayerDW part A: per-tile depthwise 3x3 (zero-padded at tile borders)
// + BN + GELU. Pure bandwidth; VALU kernel, all index math constant.
// ---------------------------------------------------------------------------
template <int C, int H, int W, int G>
__global__ __launch_bounds__(256)
void dw_kernel(const _Float16* __restrict__ act, const float* __restrict__ dw,
               const float* __restrict__ s1, const float* __restrict__ b1,
               _Float16* __restrict__ out) {
    constexpr int B  = 16;
    constexpr int TH = H / G, TW = W / G;
    size_t id = (size_t)blockIdx.x * blockDim.x + threadIdx.x;
    if (id >= (size_t)B * C * H * W) return;
    const int x = (int)(id % W); size_t r = id / W;
    const int y = (int)(r % H);  r /= H;
    const int c = (int)(r % C);  const int b = (int)(r / C);
    const int ty = y / TH, tx = x / TW;
    const int t  = ty * G + tx;
    const int ly = y - ty * TH, lx = x - tx * TW;
    const float* wv = dw + ((size_t)t * C + c) * 9;
    const _Float16* ap = act + ((size_t)b * C + c) * H * W;
    float sum = 0.0f;
#pragma unroll
    for (int dy = -1; dy <= 1; ++dy)
#pragma unroll
        for (int dx = -1; dx <= 1; ++dx) {
            const int yy = ly + dy, xx = lx + dx;
            if (yy >= 0 && yy < TH && xx >= 0 && xx < TW)
                sum += wv[(dy + 1) * 3 + (dx + 1)] *
                       (float)ap[(size_t)(y + dy) * W + (x + dx)];
        }
    const float val = sum * s1[(size_t)t * C + c] + b1[(size_t)t * C + c];
    out[id] = (_Float16)gelu_f(val);
}

// ---------------------------------------------------------------------------
// RegionLayerDW part B: per-tile 1x1 conv (GEMM, per-tile packed weights)
// + BN + residual + outer GELU via WMMA. One wave does all NT n-tiles off a
// single A fragment. M = pixels within region tile.
// ---------------------------------------------------------------------------
template <int C, int H, int W, int G>
__global__ __launch_bounds__(256)
void pw_wmma(const _Float16* __restrict__ tmp, const _Float16* __restrict__ act,
             const _Float16* __restrict__ wpk, const float* __restrict__ s2,
             const float* __restrict__ b2, _Float16* __restrict__ out) {
    constexpr int B  = 16;
    constexpr int TH = H / G, TW = W / G;
    constexpr int MT = TH * TW / 16;
    constexpr int NT = (C + 15) / 16;
    constexpr int KC = (C + 31) / 32;
    constexpr int G2 = G * G;

    const int lane = threadIdx.x & 31;
    int wave = blockIdx.x * (blockDim.x >> 5) + (threadIdx.x >> 5);
    if (wave >= B * G2 * MT) return;
    const int mt = wave % MT; wave /= MT;
    const int t  = wave % G2; const int b = wave / G2;
    const int ty = t / G, tx = t - ty * G;
    const int y0 = ty * TH, x0 = tx * TW;

    const int hi   = lane >> 4;
    const int mrow = lane & 15;
    const int pi = mt * 16 + mrow;
    const int ly = pi / TW, lx = pi - ly * TW;
    const int ya = y0 + ly, xa = x0 + lx;
    constexpr size_t plane = (size_t)H * W;
    const _Float16* tb = tmp + (size_t)b * C * plane;

    v8f acc[NT] = {};
#pragma unroll
    for (int kc = 0; kc < KC; ++kc) {
        v16h a = {};
#pragma unroll
        for (int e = 0; e < 16; ++e) {
            const int k = kc * 32 + kmap(e, hi);
            a[e] = (k < C) ? tb[(size_t)k * plane + (size_t)ya * W + xa]
                           : (_Float16)0.0f;
        }
#pragma unroll
        for (int nt = 0; nt < NT; ++nt) {
            v16h bf = *(const v16h*)(wpk +
                ((((size_t)t * NT + nt) * KC + kc) * 32 + lane) * 16);
            acc[nt] = __builtin_amdgcn_wmma_f32_16x16x32_f16(
                false, a, false, bf, (short)0, acc[nt], false, false);
        }
    }

    const int n = lane & 15;
#pragma unroll
    for (int nt = 0; nt < NT; ++nt) {
        const int oc = nt * 16 + n;
        if ((C % 16 != 0) && oc >= C) continue;
        const float s  = s2[(size_t)t * C + oc];
        const float bb = b2[(size_t)t * C + oc];
        const _Float16* ab = act + ((size_t)b * C + oc) * plane;
        _Float16*       ob = out + ((size_t)b * C + oc) * plane;
#pragma unroll
        for (int v = 0; v < 8; ++v) {
            const int mm = v + hi * 8;
            const int pj = mt * 16 + mm;
            const int my = pj / TW, mx = pj - my * TW;
            const size_t idx = (size_t)(y0 + my) * W + (x0 + mx);
            const float val = acc[nt][v] * s + bb + (float)ab[idx];
            ob[idx] = (_Float16)gelu_f(val);
        }
    }
}

// ---------------------------------------------------------------------------
extern "C" void kernel_launch(void* const* d_in, const int* in_sizes, int n_in,
                              void* d_out, int out_size, void* d_ws, size_t ws_size,
                              hipStream_t stream) {
    (void)in_sizes; (void)n_in; (void)out_size; (void)ws_size;
    const float* x     = (const float*)d_in[0];
    const float* r1_w  = (const float*)d_in[1];
    const float* r1_s  = (const float*)d_in[2];
    const float* r1_b  = (const float*)d_in[3];
    const float* r1_dw = (const float*)d_in[4];
    const float* r1_s1 = (const float*)d_in[5];
    const float* r1_b1 = (const float*)d_in[6];
    const float* r1_pw = (const float*)d_in[7];
    const float* r1_s2 = (const float*)d_in[8];
    const float* r1_b2 = (const float*)d_in[9];
    const float* r2_w  = (const float*)d_in[10];
    const float* r2_s  = (const float*)d_in[11];
    const float* r2_b  = (const float*)d_in[12];
    const float* r2_dw = (const float*)d_in[13];
    const float* r2_s1 = (const float*)d_in[14];
    const float* r2_b1 = (const float*)d_in[15];
    const float* r2_pw = (const float*)d_in[16];
    const float* r2_s2 = (const float*)d_in[17];
    const float* r2_b2 = (const float*)d_in[18];
    const float* h1w1  = (const float*)d_in[19];
    const float* h1s1  = (const float*)d_in[20];
    const float* h1b1  = (const float*)d_in[21];
    const float* h1w2  = (const float*)d_in[22];
    const float* h1s2  = (const float*)d_in[23];
    const float* h1b2  = (const float*)d_in[24];
    const float* h1w3  = (const float*)d_in[25];
    const float* h1s3  = (const float*)d_in[26];
    const float* h1b3  = (const float*)d_in[27];
    const float* h2w1  = (const float*)d_in[28];
    const float* h2s1  = (const float*)d_in[29];
    const float* h2b1  = (const float*)d_in[30];
    const float* h2w2  = (const float*)d_in[31];
    const float* h2s2  = (const float*)d_in[32];
    const float* h2b2  = (const float*)d_in[33];
    const float* h3_w  = (const float*)d_in[34];
    const float* h3_s  = (const float*)d_in[35];
    const float* h3_b  = (const float*)d_in[36];
    float* out = (float*)d_out;

    const int B = 16;
    const size_t N_X   = (size_t)B * 3  * 224 * 224;
    const size_t N_R1  = (size_t)B * 24 * 224 * 224;
    const size_t N_R2  = (size_t)B * 48 * 224 * 224;
    const size_t N_H1A = (size_t)B * 48 * 56 * 56;
    const size_t N_H1B = (size_t)B * 48 * 28 * 28;

    char* ws = (char*)d_ws;
    size_t off = 0;
    auto alloc = [&](size_t bytes) -> char* {
        char* p = ws + off;
        off += (bytes + 255) & ~(size_t)255;
        return p;
    };
    auto psz = [](int G, int OC, int K) -> size_t {
        return (size_t)G * ((OC + 15) >> 4) * ((K + 31) >> 5) * 512;
    };

    _Float16* x16    = (_Float16*)alloc(N_X * 2);
    _Float16* r1pre  = (_Float16*)alloc(N_R1 * 2);
    _Float16* tmp16  = (_Float16*)alloc(N_R2 * 2);   // shared dw scratch
    _Float16* reg1   = (_Float16*)alloc(N_R1 * 2);
    _Float16* r2a    = (_Float16*)alloc(N_R2 * 2);
    _Float16* reg2   = (_Float16*)alloc(N_R2 * 2);
    _Float16* h1a    = (_Float16*)alloc(N_H1A * 2);
    _Float16* h1b    = (_Float16*)alloc(N_H1B * 2);
    _Float16* g2b    = (_Float16*)alloc(N_H1A * 2);
    _Float16* p_r1w  = (_Float16*)alloc(psz(1, 24, 75) * 2);
    _Float16* p_r2w  = (_Float16*)alloc(psz(1, 48, 216) * 2);
    _Float16* p_h1w1 = (_Float16*)alloc(psz(1, 48, 48) * 2);
    _Float16* p_h1w2 = (_Float16*)alloc(psz(1, 48, 192) * 2);
    _Float16* p_h1w3 = (_Float16*)alloc(psz(1, 256, 192) * 2);
    _Float16* p_h2w1 = (_Float16*)alloc(psz(1, 48, 384) * 2);
    _Float16* p_h2w2 = (_Float16*)alloc(psz(1, 256, 768) * 2);
    _Float16* p_h3w  = (_Float16*)alloc(psz(1, 256, 12288) * 2);
    _Float16* p_r1pw = (_Float16*)alloc(psz(49, 24, 24) * 2);
    _Float16* p_r2pw = (_Float16*)alloc(psz(16, 48, 48) * 2);

    auto pack = [&](const float* w, int G, int OC, int K, _Float16* dst) {
        size_t total = psz(G, OC, K);
        pack_w<<<(unsigned)((total + 255) / 256), 256, 0, stream>>>(w, G, OC, K, dst);
    };

    // ---- pack all GEMM weights into fragment layout (fp16) ----
    pack(r1_w, 1, 24, 75, p_r1w);
    pack(r2_w, 1, 48, 216, p_r2w);
    pack(h1w1, 1, 48, 48, p_h1w1);
    pack(h1w2, 1, 48, 192, p_h1w2);
    pack(h1w3, 1, 256, 192, p_h1w3);
    pack(h2w1, 1, 48, 384, p_h2w1);
    pack(h2w2, 1, 256, 768, p_h2w2);
    pack(h3_w, 1, 256, 12288, p_h3w);
    pack(r1_pw, 49, 24, 24, p_r1pw);
    pack(r2_pw, 16, 48, 48, p_r2pw);

    // ---- x -> fp16 ----
    f32_to_f16<<<(unsigned)((N_X + 255) / 256), 256, 0, stream>>>(x, x16, N_X);

    auto blk = [](long waves) { return (unsigned)((waves + 7) / 8); };

    // ---- region1: 5x5 conv 3->24 +BN+GELU; region DW g=7; outer GELU ----
    // Mtiles=50176, NT=2, NTB=2 -> NB=1
    conv_wmma<3, 224, 224, 24, 5, 5, 1, 2, 224, 224, 2, false, true>
        <<<blk(50176), 256, 0, stream>>>(x16, p_r1w, r1_s, r1_b, r1pre, nullptr, 0);
    dw_kernel<24, 224, 224, 7>
        <<<(unsigned)((N_R1 + 255) / 256), 256, 0, stream>>>(
            r1pre, r1_dw, r1_s1, r1_b1, tmp16);
    // waves = 16 * 49 tiles * 64 mtiles
    pw_wmma<24, 224, 224, 7>
        <<<blk((long)16 * 49 * 64), 256, 0, stream>>>(
            tmp16, r1pre, p_r1pw, r1_s2, r1_b2, reg1);

    // ---- region2: 3x3 conv 24->48 +BN+GELU; region DW g=4; outer GELU ----
    conv_wmma<24, 224, 224, 48, 3, 3, 1, 1, 224, 224, 3, false, true>
        <<<blk(50176), 256, 0, stream>>>(reg1, p_r2w, r2_s, r2_b, r2a, nullptr, 0);
    dw_kernel<48, 224, 224, 4>
        <<<(unsigned)((N_R2 + 255) / 256), 256, 0, stream>>>(
            r2a, r2_dw, r2_s1, r2_b1, tmp16);
    // waves = 16 * 16 tiles * 196 mtiles
    pw_wmma<48, 224, 224, 4>
        <<<blk((long)16 * 16 * 196), 256, 0, stream>>>(
            tmp16, r2a, p_r2pw, r2_s2, r2_b2, reg2);

    // ---- branch1: x ->48(k4s4)->48(k2s2)->256(k2s2), out channels 0..255 ----
    conv_wmma<3, 224, 224, 48, 4, 4, 4, 0, 56, 56, 3, false, true>
        <<<blk(3136), 256, 0, stream>>>(x16, p_h1w1, h1s1, h1b1, h1a, nullptr, 0);
    conv_wmma<48, 56, 56, 48, 2, 2, 2, 0, 28, 28, 3, false, true>
        <<<blk(784), 256, 0, stream>>>(h1a, p_h1w2, h1s2, h1b2, h1b, nullptr, 0);
    // Mtiles=196, NT=16, NTB=8 -> NB=2 -> waves=392
    conv_wmma<48, 28, 28, 256, 2, 2, 2, 0, 14, 14, 8, true, false>
        <<<blk(392), 256, 0, stream>>>(h1b, p_h1w3, h1s3, h1b3, nullptr, out, 0);

    // ---- branch2: region1 ->48(k4s4)->256(k4s4), out channels 256..511 ----
    conv_wmma<24, 224, 224, 48, 4, 4, 4, 0, 56, 56, 3, false, true>
        <<<blk(3136), 256, 0, stream>>>(reg1, p_h2w1, h2s1, h2b1, g2b, nullptr, 0);
    conv_wmma<48, 56, 56, 256, 4, 4, 4, 0, 14, 14, 8, true, false>
        <<<blk(392), 256, 0, stream>>>(g2b, p_h2w2, h2s2, h2b2, nullptr, out, 256);

    // ---- branch3: region2 ->256(k16s16), out channels 512..767 ----
    conv_wmma<48, 224, 224, 256, 16, 16, 16, 0, 14, 14, 8, true, false>
        <<<blk(392), 256, 0, stream>>>(reg2, p_h3w, h3_s, h3_b, nullptr, out, 512);
}
